// PaLM_70832600645664
// MI455X (gfx1250) — matrix operand order repaired and checked by reading, not compiled, not run on hardware
//
#include <hip/hip_runtime.h>
#include <hip/hip_bf16.h>
#include <math.h>

typedef __bf16 bf16;
typedef __attribute__((ext_vector_type(16))) __bf16 v16bf;
typedef __attribute__((ext_vector_type(8)))  __bf16 v8bf;
typedef __attribute__((ext_vector_type(8)))  float  v8f;

#define DIM_     1024
#define HEADS_   16
#define DH_      64
#define DEPTH_   4
#define FFI_     4096
#define VOCAB_   32000
#define B_       4
#define N_       1024
#define FUSED_   9344     /* 1024 q | 64 k | 64 v | 8192 ff */
#define TOKENS_  (B_*N_)  /* 4096 */

// ---------------------------------------------------------------------------
// CDNA5 async memory -> LDS DMA (ASYNCcnt-tracked), per-lane 16B
// VDST operand = LDS byte address (low 32 bits of generic shared pointer)
// ---------------------------------------------------------------------------
__device__ __forceinline__ void async_load_b128(unsigned lds_off, const void* gptr) {
  asm volatile("global_load_async_to_lds_b128 %0, %1, off"
               :: "v"(lds_off), "v"(gptr)
               : "memory");
}
__device__ __forceinline__ void wait_async0() {
  asm volatile("s_wait_asynccnt 0x0" ::: "memory");
}
__device__ __forceinline__ void wait_ds0() {
  asm volatile("s_wait_dscnt 0x0" ::: "memory");
}

// ---------------------------------------------------------------------------
// Elementwise / glue kernels
// ---------------------------------------------------------------------------
__global__ void k_f32_to_bf16(const float* __restrict__ s, bf16* __restrict__ d, size_t n) {
  size_t i = (size_t)blockIdx.x * blockDim.x + threadIdx.x;
  size_t st = (size_t)gridDim.x * blockDim.x;
  for (; i < n; i += st) d[i] = (bf16)s[i];
}

__global__ void k_embed(const int* __restrict__ tok, const float* __restrict__ emb,
                        float* __restrict__ x) {
  int row = blockIdx.x;
  int t = tok[row];
  const float* src = emb + (size_t)t * DIM_;
  float* dst = x + (size_t)row * DIM_;
  for (int c = threadIdx.x; c < DIM_; c += blockDim.x) dst[c] = src[c];
}

__global__ void k_layernorm_bf16(const float* __restrict__ x, const float* __restrict__ g,
                                 bf16* __restrict__ y) {
  __shared__ float red[256];
  int row = blockIdx.x, tid = threadIdx.x;
  const float* xr = x + (size_t)row * DIM_;
  float s = 0.f;
  for (int c = tid; c < DIM_; c += 256) s += xr[c];
  red[tid] = s; __syncthreads();
  for (int k = 128; k > 0; k >>= 1) { if (tid < k) red[tid] += red[tid + k]; __syncthreads(); }
  float mu = red[0] * (1.0f / DIM_);
  __syncthreads();
  float v = 0.f;
  for (int c = tid; c < DIM_; c += 256) { float d = xr[c] - mu; v += d * d; }
  red[tid] = v; __syncthreads();
  for (int k = 128; k > 0; k >>= 1) { if (tid < k) red[tid] += red[tid + k]; __syncthreads(); }
  float inv = rsqrtf(red[0] * (1.0f / DIM_) + 1e-5f);
  bf16* yr = y + (size_t)row * DIM_;
  for (int c = tid; c < DIM_; c += 256) yr[c] = (bf16)((xr[c] - mu) * inv * g[c]);
}

// RoPE with xpos scaling; emits bf16 q[b,h,n,d], k[b,n,d], v[b,n,d]
__global__ void k_rope(const float* __restrict__ proj, bf16* __restrict__ qo,
                       bf16* __restrict__ ko, bf16* __restrict__ vo) {
  int row = blockIdx.x;                 // b*N + t
  int b = row >> 10, t = row & 1023;
  const float* pr = proj + (size_t)row * FUSED_;
  int tid = threadIdx.x;
  for (int e = tid; e < HEADS_ * DH_; e += 256) {
    int h = e >> 6, d = e & 63, j = d & 31;
    float invf = __powf(10000.0f, -(float)(2 * j) * (1.0f / DH_));
    float fr = (float)t * invf;
    float cs = __cosf(fr), sn = __sinf(fr);
    float base = ((float)(2 * j) + 0.4f * DH_) / (1.4f * DH_);
    float pw = ((float)t - 512.0f) * (1.0f / 512.0f);
    float sc = __powf(base, pw);
    float qv = pr[h * DH_ + d];
    float rh = (d < 32) ? -pr[h * DH_ + d + 32] : pr[h * DH_ + d - 32];
    float q = 0.125f * (qv * cs + rh * sn) * sc;   // DH^-0.5 = 0.125
    qo[((size_t)(b * HEADS_ + h) * N_ + t) * DH_ + d] = (bf16)q;
  }
  if (tid < DH_) {
    int d = tid, j = d & 31;
    float invf = __powf(10000.0f, -(float)(2 * j) * (1.0f / DH_));
    float fr = (float)t * invf;
    float cs = __cosf(fr), sn = __sinf(fr);
    float base = ((float)(2 * j) + 0.4f * DH_) / (1.4f * DH_);
    float pw = ((float)t - 512.0f) * (1.0f / 512.0f);
    float sc = __powf(base, pw);
    float kv = pr[1024 + d];
    float rh = (d < 32) ? -pr[1024 + d + 32] : pr[1024 + d - 32];
    ko[(size_t)row * DH_ + d] = (bf16)((kv * cs + rh * sn) / sc);
    vo[(size_t)row * DH_ + d] = (bf16)pr[1088 + d];
  }
}

__global__ void k_softmax_causal(float* __restrict__ sim, bf16* __restrict__ attn) {
  __shared__ float red[256];
  int rowIdx = blockIdx.x;              // (b*H+h)*N + i
  int i = rowIdx & 1023;
  float* row = sim + (size_t)rowIdx * N_;
  bf16* arow = attn + (size_t)rowIdx * N_;
  int tid = threadIdx.x;
  float m = -3.4e38f;
  for (int j = tid; j <= i; j += 256) m = fmaxf(m, row[j]);
  red[tid] = m; __syncthreads();
  for (int k = 128; k > 0; k >>= 1) { if (tid < k) red[tid] = fmaxf(red[tid], red[tid + k]); __syncthreads(); }
  m = red[0]; __syncthreads();
  float s = 0.f;
  for (int j = tid; j <= i; j += 256) { float e = __expf(row[j] - m); row[j] = e; s += e; }
  red[tid] = s; __syncthreads();
  for (int k = 128; k > 0; k >>= 1) { if (tid < k) red[tid] += red[tid + k]; __syncthreads(); }
  float inv = 1.0f / red[0];
  for (int j = tid; j < N_; j += 256)
    arow[j] = (j <= i) ? (bf16)(row[j] * inv) : (bf16)0.0f;
}

// out[b,h,n,d] fp32 -> ctx[b,n, h*DH+d] bf16
__global__ void k_ctx_permute(const float* __restrict__ o, bf16* __restrict__ ctx, size_t n) {
  size_t i = (size_t)blockIdx.x * blockDim.x + threadIdx.x;
  size_t st = (size_t)gridDim.x * blockDim.x;
  for (; i < n; i += st) {
    int c = (int)(i & 1023);
    size_t row = i >> 10;               // b*N + t
    int b = (int)(row >> 10), t = (int)(row & 1023);
    int h = c >> 6, d = c & 63;
    ctx[i] = (bf16)o[(((size_t)(b * HEADS_ + h)) * N_ + t) * DH_ + d];
  }
}

__global__ void k_silu_gate(const float* __restrict__ proj, bf16* __restrict__ out, size_t n) {
  size_t i = (size_t)blockIdx.x * blockDim.x + threadIdx.x;
  size_t st = (size_t)gridDim.x * blockDim.x;
  for (; i < n; i += st) {
    size_t row = i >> 12;               // / 4096
    int c = (int)(i & (FFI_ - 1));
    float val = proj[row * FUSED_ + 1152 + c];
    float g   = proj[row * FUSED_ + 1152 + FFI_ + c];
    float sig = 1.0f / (1.0f + __expf(-g));
    out[i] = (bf16)(g * sig * val);
  }
}

__global__ void k_residual(float* __restrict__ x, const float* __restrict__ a,
                           const float* __restrict__ f, size_t n) {
  size_t i = (size_t)blockIdx.x * blockDim.x + threadIdx.x;
  size_t st = (size_t)gridDim.x * blockDim.x;
  for (; i < n; i += st) x[i] += a[i] + f[i];
}

// ---------------------------------------------------------------------------
// WMMA bf16 GEMM: C[M,N] (f32) = A[M,K] (bf16 row-major) * B (bf16)
//   TRANSB=false: B stored K x N row-major (in-LDS transpose, scalar staging)
//   TRANSB=true : B stored N x K row-major (async b128 DMA staging)
// Block 256 thr (8 waves), tile 128 x BN, wave tile 32 x BN/2
// (BN=128 -> 8 WMMA / wave / K-step). Double-buffered LDS, async global->LDS,
// one barrier per K-step.
// Batched via blockIdx.z; B advances every `bdiv` batches (multi-query share).
// ---------------------------------------------------------------------------
template <bool TRANSB, int BN>
__global__ __launch_bounds__(256) void k_wmma_gemm(
    const bf16* __restrict__ A, const bf16* __restrict__ B, float* __restrict__ C,
    int M, int N, int K, long long sA_, long long sB_, long long sC_, int bdiv) {
  constexpr int NT = BN / 32;           // WMMA tiles per wave along N
  __shared__ __attribute__((aligned(16))) bf16 sA[2][128][40];
  __shared__ __attribute__((aligned(16))) bf16 sB[2][BN][40];
  A += (size_t)blockIdx.z * sA_;
  B += (size_t)(blockIdx.z / bdiv) * sB_;
  C += (size_t)blockIdx.z * sC_;
  const int tid  = threadIdx.x;
  const int lane = tid & 31, wave = tid >> 5;
  const int wm = (wave & 3) * 32, wn = (wave >> 2) * (BN / 2);
  const int lmod = lane & 15, lhalf = lane >> 4;
  const int m0 = blockIdx.y * 128, n0 = blockIdx.x * BN;

  auto stage = [&](int k0, int buf) {
    // A tile 128x32: async DMA, two b128 chunks per thread
    {
      int c = tid;
#pragma unroll
      for (int i = 0; i < 2; ++i, c += 256) {
        int r = c >> 2, c8 = (c & 3) * 8;
        async_load_b128((unsigned)(size_t)&sA[buf][r][c8],
                        &A[(size_t)(m0 + r) * K + k0 + c8]);
      }
    }
    if (TRANSB) {
      // B stored NxK: async DMA b128 chunks straight into sB[n][k]
#pragma unroll
      for (int i = 0; i < BN / 64; ++i) {
        int c = tid + i * 256;
        int r = c >> 2, c8 = (c & 3) * 8;
        async_load_b128((unsigned)(size_t)&sB[buf][r][c8],
                        &B[(size_t)(n0 + r) * K + k0 + c8]);
      }
    } else {
      // B stored KxN: transpose into sB[n][k] (coalesced along n)
#pragma unroll
      for (int i = 0; i < BN / 8; ++i) {
        int e = tid + i * 256;
        int kk = e / BN, nn = e % BN;
        sB[buf][nn][kk] = B[(size_t)(k0 + kk) * N + n0 + nn];
      }
    }
  };

  v8f acc[2][NT] = {};
  const int nk = K / 32;

  stage(0, 0);
  wait_async0();
  wait_ds0();
  __syncthreads();

  for (int ki = 0; ki < nk; ++ki) {
    const int cur = ki & 1;
    if (ki + 1 < nk) stage((ki + 1) * 32, cur ^ 1);   // overlap DMA with math

    v16bf afr[2], bfr[NT];
#pragma unroll
    for (int t = 0; t < 2; ++t) {
      // A fragment: lane lmod = row m; K = lhalf*8 + [0..7] and 16+lhalf*8 + [0..7]
      int m = wm + t * 16 + lmod;
      v8bf alo = *(const v8bf*)&sA[cur][m][lhalf * 8];
      v8bf ahi = *(const v8bf*)&sA[cur][m][16 + lhalf * 8];
      afr[t] = __builtin_shufflevector(alo, ahi, 0, 1, 2, 3, 4, 5, 6, 7,
                                       8, 9, 10, 11, 12, 13, 14, 15);
    }
#pragma unroll
    for (int t = 0; t < NT; ++t) {
      // B fragment: lane lmod = col n; K = lhalf*8 + [0..7] and 16+lhalf*8 + [0..7]
      int n = wn + t * 16 + lmod;
      v8bf blo = *(const v8bf*)&sB[cur][n][lhalf * 8];
      v8bf bhi = *(const v8bf*)&sB[cur][n][16 + lhalf * 8];
      bfr[t] = __builtin_shufflevector(blo, bhi, 0, 1, 2, 3, 4, 5, 6, 7,
                                       8, 9, 10, 11, 12, 13, 14, 15);
    }
#pragma unroll
    for (int tm = 0; tm < 2; ++tm)
#pragma unroll
      for (int tn = 0; tn < NT; ++tn)
        acc[tm][tn] = __builtin_amdgcn_wmma_f32_16x16x32_bf16(
            false, afr[tm], false, bfr[tn], (short)0, acc[tm][tn], false, false);

    wait_ds0();      // this wave's fragment reads done (buffer reused next iter)
    wait_async0();   // this wave's DMA for next buffer landed
    __syncthreads(); // whole block in lockstep: one barrier per K-step
  }

  // C/D layout: VGPR r -> m = tile_m + lhalf*8 + r ; n = tile_n + lmod
#pragma unroll
  for (int tm = 0; tm < 2; ++tm) {
    int mb = m0 + wm + tm * 16 + lhalf * 8;
#pragma unroll
    for (int tn = 0; tn < NT; ++tn) {
      int n = n0 + wn + tn * 16 + lmod;
#pragma unroll
      for (int r = 0; r < 8; ++r)
        C[(size_t)(mb + r) * N + n] = acc[tm][tn][r];
    }
  }
}

// ---------------------------------------------------------------------------
extern "C" void kernel_launch(void* const* d_in, const int* in_sizes, int n_in,
                              void* d_out, int out_size, void* d_ws, size_t ws_size,
                              hipStream_t stream) {
  (void)in_sizes; (void)n_in; (void)out_size; (void)ws_size;
  const int*   tokens  = (const int*)d_in[0];
  const float* emb     = (const float*)d_in[1];
  const float* gblocks = (const float*)d_in[2];
  const float* wfused  = (const float*)d_in[3];
  const float* wattn   = (const float*)d_in[4];
  const float* wff     = (const float*)d_in[5];
  const float* gfinal  = (const float*)d_in[6];
  float* out = (float*)d_out;

  char* p = (char*)d_ws;
  auto alloc = [&](size_t bytes) -> void* {
    void* r = (void*)p;
    p += (bytes + 255) & ~(size_t)255;
    return r;
  };

  const size_t NWF = (size_t)DEPTH_ * DIM_ * FUSED_;
  const size_t NWA = (size_t)DEPTH_ * DIM_ * DIM_;
  const size_t NWO = (size_t)DEPTH_ * FFI_ * DIM_;
  const size_t NEM = (size_t)VOCAB_ * DIM_;

  bf16*  wfused_bf = (bf16*)alloc(NWF * 2);
  bf16*  wattn_bf  = (bf16*)alloc(NWA * 2);
  bf16*  wff_bf    = (bf16*)alloc(NWO * 2);
  bf16*  emb_bf    = (bf16*)alloc(NEM * 2);
  float* x         = (float*)alloc((size_t)TOKENS_ * DIM_ * 4);
  bf16*  hnorm_bf  = (bf16*)alloc((size_t)TOKENS_ * DIM_ * 2);
  float* proj      = (float*)alloc((size_t)TOKENS_ * FUSED_ * 4);
  bf16*  q_bf      = (bf16*)alloc((size_t)B_ * HEADS_ * N_ * DH_ * 2);
  bf16*  k_bf      = (bf16*)alloc((size_t)B_ * N_ * DH_ * 2);
  bf16*  v_bf      = (bf16*)alloc((size_t)B_ * N_ * DH_ * 2);
  float* sim       = (float*)alloc((size_t)B_ * HEADS_ * N_ * N_ * 4);
  bf16*  attn_bf   = (bf16*)alloc((size_t)B_ * HEADS_ * N_ * N_ * 2);
  float* obhnd     = (float*)alloc((size_t)B_ * HEADS_ * N_ * DH_ * 4);
  bf16*  ctx_bf    = (bf16*)alloc((size_t)TOKENS_ * DIM_ * 2);
  float* attnout   = (float*)alloc((size_t)TOKENS_ * DIM_ * 4);
  bf16*  ffact_bf  = (bf16*)alloc((size_t)TOKENS_ * FFI_ * 2);
  float* ffout     = (float*)alloc((size_t)TOKENS_ * DIM_ * 4);
  bf16*  fin_bf    = (bf16*)alloc((size_t)TOKENS_ * DIM_ * 2);

  auto cvt = [&](const float* s, bf16* d, size_t n) {
    int blocks = (int)((n + 255) / 256);
    k_f32_to_bf16<<<blocks, 256, 0, stream>>>(s, d, n);
  };
  auto gemm = [&](const bf16* Am, const bf16* Bm, float* Cm, int M, int Nn, int K,
                  int batch, long long sa, long long sb, long long sc, int bdiv,
                  bool transB) {
    if ((Nn & 127) == 0) {
      dim3 grid(Nn / 128, M / 128, batch);
      if (transB)
        k_wmma_gemm<true, 128><<<grid, 256, 0, stream>>>(Am, Bm, Cm, M, Nn, K, sa, sb, sc, bdiv);
      else
        k_wmma_gemm<false, 128><<<grid, 256, 0, stream>>>(Am, Bm, Cm, M, Nn, K, sa, sb, sc, bdiv);
    } else {
      dim3 grid(Nn / 64, M / 128, batch);   // only the N=64 attn@V GEMM (non-trans)
      k_wmma_gemm<false, 64><<<grid, 256, 0, stream>>>(Am, Bm, Cm, M, Nn, K, sa, sb, sc, bdiv);
    }
  };

  // One-time weight precision conversion (graph-captured each call; deterministic)
  cvt(wfused, wfused_bf, NWF);
  cvt(wattn,  wattn_bf,  NWA);
  cvt(wff,    wff_bf,    NWO);
  cvt(emb,    emb_bf,    NEM);

  // x = emb[tokens]
  k_embed<<<TOKENS_, 256, 0, stream>>>(tokens, emb, x);

  for (int l = 0; l < DEPTH_; ++l) {
    // h = layernorm(x) -> bf16
    k_layernorm_bf16<<<TOKENS_, 256, 0, stream>>>(x, gblocks + (size_t)l * DIM_, hnorm_bf);
    // proj = h @ Wf   [4096 x 9344]
    gemm(hnorm_bf, wfused_bf + (size_t)l * DIM_ * FUSED_, proj,
         TOKENS_, FUSED_, DIM_, 1, 0, 0, 0, 1, false);
    // RoPE/xpos -> q,k + v passthrough (bf16)
    k_rope<<<TOKENS_, 256, 0, stream>>>(proj, q_bf, k_bf, v_bf);
    // sim[b,h] = q[b,h] @ k[b]^T  (k stored NxK -> TRANSB path)
    gemm(q_bf, k_bf, sim, N_, N_, DH_, B_ * HEADS_,
         (long long)N_ * DH_, (long long)N_ * DH_, (long long)N_ * N_, HEADS_, true);
    // causal softmax -> bf16 probs
    k_softmax_causal<<<B_ * HEADS_ * N_, 256, 0, stream>>>(sim, attn_bf);
    // out[b,h] = attn[b,h] @ v[b]
    gemm(attn_bf, v_bf, obhnd, N_, DH_, N_, B_ * HEADS_,
         (long long)N_ * N_, (long long)N_ * DH_, (long long)N_ * DH_, HEADS_, false);
    // [b,h,n,d] -> [b,n,h*d] bf16
    k_ctx_permute<<<4096, 256, 0, stream>>>(obhnd, ctx_bf, (size_t)TOKENS_ * DIM_);
    // attn_out = ctx @ Wo
    gemm(ctx_bf, wattn_bf + (size_t)l * DIM_ * DIM_, attnout,
         TOKENS_, DIM_, DIM_, 1, 0, 0, 0, 1, false);
    // ff activation: silu(gate) * val -> bf16
    k_silu_gate<<<4096, 256, 0, stream>>>(proj, ffact_bf, (size_t)TOKENS_ * FFI_);
    // ff_out = act @ Wff
    gemm(ffact_bf, wff_bf + (size_t)l * FFI_ * DIM_, ffout,
         TOKENS_, DIM_, FFI_, 1, 0, 0, 0, 1, false);
    // x = x + attn_out + ff_out
    k_residual<<<4096, 256, 0, stream>>>(x, attnout, ffout, (size_t)TOKENS_ * DIM_);
  }

  // final layernorm + tied logits: out = ln(x) @ emb^T  (emb is VOCAB x DIM = NxK)
  k_layernorm_bf16<<<TOKENS_, 256, 0, stream>>>(x, gfinal, fin_bf);
  gemm(fin_bf, emb_bf, out, TOKENS_, VOCAB_, DIM_, 1, 0, 0, 0, 1, true);
}
